// Decoder_90744069030122
// MI455X (gfx1250) — compile-verified, matrix operand order checked by reading, not run on hardware
//
#include <hip/hip_runtime.h>
#include <hip/hip_bf16.h>

typedef __attribute__((ext_vector_type(16))) _Float16 v16h;
typedef __attribute__((ext_vector_type(8)))  _Float16 v8h;
typedef __attribute__((ext_vector_type(8)))  float    v8f;

#define H      768
#define L      60
#define NL     4
#define NHEAD  4
#define DH     192     // H / NHEAD
#define FF     512
#define SLOT   128
#define INTENT 22
#define BATCH  256
#define MROWS  (BATCH * L)   // 15360
#define LP     64            // L padded to WMMA multiple

// ---------------------------------------------------------------------------
// helpers: convert 16 fp32 (from 4 float4 loads) into two packed v8h
// ---------------------------------------------------------------------------
__device__ __forceinline__ void cvt16_store(_Float16* dst, const float* src, float msk) {
  float4 f0 = *(const float4*)(src + 0);
  float4 f1 = *(const float4*)(src + 4);
  float4 f2 = *(const float4*)(src + 8);
  float4 f3 = *(const float4*)(src + 12);
  v8h h0, h1;
  h0[0] = (_Float16)(f0.x * msk); h0[1] = (_Float16)(f0.y * msk);
  h0[2] = (_Float16)(f0.z * msk); h0[3] = (_Float16)(f0.w * msk);
  h0[4] = (_Float16)(f1.x * msk); h0[5] = (_Float16)(f1.y * msk);
  h0[6] = (_Float16)(f1.z * msk); h0[7] = (_Float16)(f1.w * msk);
  h1[0] = (_Float16)(f2.x * msk); h1[1] = (_Float16)(f2.y * msk);
  h1[2] = (_Float16)(f2.z * msk); h1[3] = (_Float16)(f2.w * msk);
  h1[4] = (_Float16)(f3.x * msk); h1[5] = (_Float16)(f3.y * msk);
  h1[6] = (_Float16)(f3.z * msk); h1[7] = (_Float16)(f3.w * msk);
  *(v8h*)(dst + 0) = h0;
  *(v8h*)(dst + 8) = h1;
}

__device__ __forceinline__ v16h frag_from_lds(const _Float16* base, int kb) {
  v8h lo = *(const v8h*)(base + kb);
  v8h hi = *(const v8h*)(base + 16 + kb);
  v16h a;
  #pragma unroll
  for (int j = 0; j < 8; ++j) { a[j] = lo[j]; a[8 + j] = hi[j]; }
  return a;
}

// ---------------------------------------------------------------------------
// x[b,l,:] = emb[input[b,l]] * sqrt(H) + pe[b,:]   (PE indexed by batch, as in ref)
// ---------------------------------------------------------------------------
__global__ void prep_kernel(const int* __restrict__ inp, const float* __restrict__ emb,
                            float* __restrict__ x) {
  int row = blockIdx.x;            // b*L + l
  int b   = row / L;
  int idx = inp[row];
  const float sq = 27.712812921102035f;            // sqrt(768)
  const float c  = -9.210340371976184f / (float)H; // -ln(10000)/H
  for (int h = threadIdx.x; h < H; h += blockDim.x) {
    float div = __expf((float)(h & ~1) * c);
    float ang = (float)b * div;
    float pe  = (h & 1) ? __cosf(ang) : __sinf(ang);
    x[(size_t)row * H + h] = emb[(size_t)idx * H + h] * sq + pe;
  }
}

// ---------------------------------------------------------------------------
// Stable left-pack rows of enc where mask==1; unused rows zero.
// ---------------------------------------------------------------------------
__global__ void compact_kernel(const float* __restrict__ enc, const int* __restrict__ mask,
                               float* __restrict__ newt) {
  __shared__ int dest[L];
  int b = blockIdx.x;
  if (threadIdx.x == 0) {
    int c = 0;
    for (int l = 0; l < L; ++l) {
      if (mask[b * L + l] == 1) dest[l] = c++;
      else                      dest[l] = -1;
    }
  }
  __syncthreads();
  for (int i = threadIdx.x; i < L * H; i += blockDim.x)
    newt[(size_t)b * L * H + i] = 0.f;
  __syncthreads();
  for (int l = 0; l < L; ++l) {
    int d = dest[l];
    if (d >= 0) {
      for (int h = threadIdx.x; h < H; h += blockDim.x)
        newt[((size_t)b * L + d) * H + h] = enc[((size_t)b * L + l) * H + h];
    }
  }
}

// ---------------------------------------------------------------------------
// Intent head: z = einsum('lh,ml->hm'), relu, mean over m, logits, log_softmax.
// ---------------------------------------------------------------------------
__global__ void intent_kernel(const float* __restrict__ enc, const float* __restrict__ sqW,
                              const float* __restrict__ sqb, const float* __restrict__ iW,
                              const float* __restrict__ ib, float* __restrict__ out) {
  const int MQ = L / 4;  // 15
  __shared__ float Ws[(L / 4) * L];
  __shared__ float pooled[H];
  __shared__ float lg[INTENT];
  __shared__ float red2[2];
  int b = blockIdx.x;
  for (int i = threadIdx.x; i < MQ * L; i += blockDim.x) Ws[i] = sqW[i];
  __syncthreads();
  for (int h = threadIdx.x; h < H; h += blockDim.x) {
    float z[15];
    #pragma unroll
    for (int m = 0; m < 15; ++m) z[m] = sqb[m];
    for (int l = 0; l < L; ++l) {
      float e = enc[((size_t)b * L + l) * H + h];
      #pragma unroll
      for (int m = 0; m < 15; ++m) z[m] += e * Ws[m * L + l];
    }
    float s = 0.f;
    #pragma unroll
    for (int m = 0; m < 15; ++m) s += fmaxf(z[m], 0.f);
    pooled[h] = s * (1.f / 15.f);
  }
  __syncthreads();
  if (threadIdx.x < INTENT) {
    float s = ib[threadIdx.x];
    for (int h = 0; h < H; ++h) s += pooled[h] * iW[(size_t)threadIdx.x * H + h];
    lg[threadIdx.x] = s;
  }
  __syncthreads();
  if (threadIdx.x == 0) {
    float mx = -1e30f;
    for (int i = 0; i < INTENT; ++i) mx = fmaxf(mx, lg[i]);
    float sm = 0.f;
    for (int i = 0; i < INTENT; ++i) sm += __expf(lg[i] - mx);
    red2[0] = mx; red2[1] = logf(sm);
  }
  __syncthreads();
  if (threadIdx.x < INTENT)
    out[(size_t)b * INTENT + threadIdx.x] = lg[threadIdx.x] - red2[0] - red2[1];
}

// ---------------------------------------------------------------------------
// WMMA GEMM: Y[M,N] = act(X[M,K] @ W[N,K]^T + bias[N]).  M%64==0, N%64==0, K%32==0.
// 128 threads = 4 waves; block tile 64x64; K panel 32 staged in LDS as f16.
// ---------------------------------------------------------------------------
template <int ACT>  // 0 = none, 1 = relu
__global__ __launch_bounds__(128) void gemm_wmma(const float* __restrict__ X,
                                                 const float* __restrict__ W,
                                                 const float* __restrict__ bias,
                                                 float* __restrict__ Y,
                                                 int M, int N, int K) {
  __shared__ __align__(16) _Float16 As[64 * 40];
  __shared__ __align__(16) _Float16 Bs[64 * 40];
  int n0 = blockIdx.x * 64;
  int m0 = blockIdx.y * 64;
  int tid  = threadIdx.x;
  int wave = tid >> 5;
  int lane = tid & 31;
  int lr   = lane & 15;
  int kb   = (lane >> 4) * 8;
  v8f acc[4] = {};
  int ldr = tid >> 1;         // 0..63
  int ldc = (tid & 1) * 16;   // 0 or 16
  for (int k0 = 0; k0 < K; k0 += 32) {
    cvt16_store(&As[ldr * 40 + ldc], X + (size_t)(m0 + ldr) * K + k0 + ldc, 1.f);
    cvt16_store(&Bs[ldr * 40 + ldc], W + (size_t)(n0 + ldr) * K + k0 + ldc, 1.f);
    __syncthreads();
    v16h a = frag_from_lds(&As[(wave * 16 + lr) * 40], kb);
    #pragma unroll
    for (int t = 0; t < 4; ++t) {
      v16h bf = frag_from_lds(&Bs[(t * 16 + lr) * 40], kb);
      acc[t] = __builtin_amdgcn_wmma_f32_16x16x32_f16(false, a, false, bf,
                                                      (short)0, acc[t], false, false);
    }
    __syncthreads();
  }
  int rbase = m0 + wave * 16 + (lane >> 4) * 8;
  #pragma unroll
  for (int t = 0; t < 4; ++t) {
    int col = n0 + t * 16 + lr;
    float bv = bias[col];
    #pragma unroll
    for (int v = 0; v < 8; ++v) {
      float val = acc[t][v] + bv;
      if (ACT == 1) val = fmaxf(val, 0.f);
      Y[(size_t)(rbase + v) * N + col] = val;
    }
  }
}

// ---------------------------------------------------------------------------
// Self-attention, one block per (batch, head). L padded 60->64 (zero-filled).
// S = QK^T/sqrt(dh) + causal mask; softmax; O = P V.  All matmuls via WMMA.
// All global->LDS staging is branch-free (clamped row + 0/1 multiplier).
// ---------------------------------------------------------------------------
__global__ __launch_bounds__(128) void attn_kernel(const float* __restrict__ Q,
                                                   const float* __restrict__ Kt,
                                                   const float* __restrict__ V,
                                                   float* __restrict__ O) {
  __shared__ __align__(16) _Float16 Qs[64 * 40];
  __shared__ __align__(16) _Float16 Ks[64 * 40];
  __shared__ __align__(16) _Float16 Vt[DH * LP];  // Vt[c][k] = V[k][c]
  __shared__ float Sf[LP * LP];
  int blk = blockIdx.x;
  int b = blk / NHEAD, h = blk % NHEAD;
  int bL = b * L, hc = h * DH;
  int tid = threadIdx.x, wave = tid >> 5, lane = tid & 31;
  int lr = lane & 15, kb = (lane >> 4) * 8;
  int ldr = tid >> 1;          // 0..63
  int ldc = (tid & 1) * 16;    // 0 or 16
  int srow = (ldr < L) ? ldr : 0;      // clamped source row
  float smsk = (ldr < L) ? 1.f : 0.f;  // zero-fill pad rows

  // V transposed into LDS, branch-free
  for (int i = tid; i < DH * LP; i += 128) {
    int c = i >> 6;
    int k = i & 63;
    int ks_ = (k < L) ? k : 0;
    float m = (k < L) ? 1.f : 0.f;
    Vt[c * LP + k] = (_Float16)(V[((size_t)(bL + ks_)) * H + hc + c] * m);
  }

  // ---- S = Q K^T over DH in panels of 32, staged through LDS ----
  v8f acc[4] = {};
  int ar = wave * 16 + lr;
  for (int k0 = 0; k0 < DH; k0 += 32) {
    cvt16_store(&Qs[ldr * 40 + ldc], Q  + (size_t)(bL + srow) * H + hc + k0 + ldc, smsk);
    cvt16_store(&Ks[ldr * 40 + ldc], Kt + (size_t)(bL + srow) * H + hc + k0 + ldc, smsk);
    __syncthreads();
    v16h a = frag_from_lds(&Qs[ar * 40], kb);
    #pragma unroll
    for (int t = 0; t < 4; ++t) {
      v16h bf = frag_from_lds(&Ks[(t * 16 + lr) * 40], kb);
      acc[t] = __builtin_amdgcn_wmma_f32_16x16x32_f16(false, a, false, bf,
                                                      (short)0, acc[t], false, false);
    }
    __syncthreads();
  }
  const float scale = 0.07216878364870323f;  // 1/sqrt(192)
  int rbase = wave * 16 + (lane >> 4) * 8;
  #pragma unroll
  for (int t = 0; t < 4; ++t) {
    int c = t * 16 + lr;
    #pragma unroll
    for (int v = 0; v < 8; ++v) {
      int r = rbase + v;
      float val = acc[t][v] * scale;
      if (c > r || c >= L) val = -1e30f;   // causal + pad mask
      Sf[r * LP + c] = val;
    }
  }
  __syncthreads();
  // ---- row softmax (64 rows, one thread per row) ----
  if (tid < LP) {
    float mx = -1e30f;
    for (int c = 0; c < LP; ++c) mx = fmaxf(mx, Sf[tid * LP + c]);
    float sm = 0.f;
    for (int c = 0; c < LP; ++c) sm += __expf(Sf[tid * LP + c] - mx);
    float inv = 1.f / sm;
    for (int c = 0; c < LP; ++c) Sf[tid * LP + c] = __expf(Sf[tid * LP + c] - mx) * inv;
  }
  __syncthreads();
  // ---- O = P @ V ----
  #pragma unroll 1
  for (int n = 0; n < DH / 16; ++n) {
    v8f oacc = {};
    #pragma unroll
    for (int ks2 = 0; ks2 < 2; ++ks2) {
      v16h a;
      #pragma unroll
      for (int j = 0; j < 8; ++j) {
        a[j]     = (_Float16)Sf[ar * LP + ks2 * 32 + kb + j];
        a[8 + j] = (_Float16)Sf[ar * LP + ks2 * 32 + 16 + kb + j];
      }
      v16h bf = frag_from_lds(&Vt[(n * 16 + lr) * LP + ks2 * 32], kb);
      oacc = __builtin_amdgcn_wmma_f32_16x16x32_f16(false, a, false, bf,
                                                    (short)0, oacc, false, false);
    }
    int col = hc + n * 16 + lr;
    #pragma unroll
    for (int v = 0; v < 8; ++v) {
      int r = rbase + v;
      if (r < L) O[(size_t)(bL + r) * H + col] = oacc[v];
    }
  }
}

// ---------------------------------------------------------------------------
// x = LayerNorm(x + y) * g + b   (one block per row of 768)
// ---------------------------------------------------------------------------
__global__ __launch_bounds__(256) void ln_res_kernel(float* __restrict__ x,
                                                     const float* __restrict__ y,
                                                     const float* __restrict__ g,
                                                     const float* __restrict__ bb) {
  __shared__ float red[256];
  size_t row = blockIdx.x;
  int tid = threadIdx.x;
  float v0 = x[row * H + tid]       + y[row * H + tid];
  float v1 = x[row * H + tid + 256] + y[row * H + tid + 256];
  float v2 = x[row * H + tid + 512] + y[row * H + tid + 512];
  red[tid] = v0 + v1 + v2;
  __syncthreads();
  for (int s = 128; s > 0; s >>= 1) { if (tid < s) red[tid] += red[tid + s]; __syncthreads(); }
  float mean = red[0] * (1.f / (float)H);
  __syncthreads();
  float d0 = v0 - mean, d1 = v1 - mean, d2 = v2 - mean;
  red[tid] = d0 * d0 + d1 * d1 + d2 * d2;
  __syncthreads();
  for (int s = 128; s > 0; s >>= 1) { if (tid < s) red[tid] += red[tid + s]; __syncthreads(); }
  float inv = rsqrtf(red[0] * (1.f / (float)H) + 1e-5f);
  x[row * H + tid]       = d0 * inv * g[tid]       + bb[tid];
  x[row * H + tid + 256] = d1 * inv * g[tid + 256] + bb[tid + 256];
  x[row * H + tid + 512] = d2 * inv * g[tid + 512] + bb[tid + 512];
}

// ---------------------------------------------------------------------------
// Row-wise log_softmax over 128 slots.
// ---------------------------------------------------------------------------
__global__ __launch_bounds__(128) void slot_ls_kernel(const float* __restrict__ logits,
                                                      float* __restrict__ out) {
  __shared__ float red[128];
  size_t row = blockIdx.x;
  int tid = threadIdx.x;
  float v = logits[row * SLOT + tid];
  red[tid] = v;
  __syncthreads();
  for (int s = 64; s > 0; s >>= 1) { if (tid < s) red[tid] = fmaxf(red[tid], red[tid + s]); __syncthreads(); }
  float mx = red[0];
  __syncthreads();
  red[tid] = __expf(v - mx);
  __syncthreads();
  for (int s = 64; s > 0; s >>= 1) { if (tid < s) red[tid] += red[tid + s]; __syncthreads(); }
  float ls = logf(red[0]);
  out[row * SLOT + tid] = v - mx - ls;
}

// ---------------------------------------------------------------------------
extern "C" void kernel_launch(void* const* d_in, const int* in_sizes, int n_in,
                              void* d_out, int out_size, void* d_ws, size_t ws_size,
                              hipStream_t stream) {
  (void)in_sizes; (void)n_in; (void)out_size; (void)ws_size;
  const int*   input  = (const int*)  d_in[0];
  const float* enc    = (const float*)d_in[1];
  /* d_in[2] encoder_maskings: unused by reference forward */
  const int*   subtok = (const int*)  d_in[3];
  const float* emb    = (const float*)d_in[4];
  const float* sqW    = (const float*)d_in[5];
  const float* sqb    = (const float*)d_in[6];
  const float* iW     = (const float*)d_in[7];
  const float* ib     = (const float*)d_in[8];
  const float* slW    = (const float*)d_in[9];
  const float* slb    = (const float*)d_in[10];
  const float* saWq   = (const float*)d_in[11];
  const float* saWk   = (const float*)d_in[12];
  const float* saWv   = (const float*)d_in[13];
  const float* saWo   = (const float*)d_in[14];
  const float* sab    = (const float*)d_in[15];
  const float* caWv   = (const float*)d_in[18];
  const float* caWo   = (const float*)d_in[19];
  const float* cab    = (const float*)d_in[20];
  const float* fW1    = (const float*)d_in[21];
  const float* fb1    = (const float*)d_in[22];
  const float* fW2    = (const float*)d_in[23];
  const float* fb2    = (const float*)d_in[24];
  const float* lng    = (const float*)d_in[25];
  const float* lnb    = (const float*)d_in[26];

  float* ws = (float*)d_ws;
  const size_t T = (size_t)MROWS * H;  // 11,796,480 floats
  float* x    = ws;
  float* newt = ws + T;
  float* bufA = ws + 2 * T;
  float* bufB = ws + 3 * T;
  float* bufC = ws + 4 * T;
  float* bufD = ws + 5 * T;
  float* out  = (float*)d_out;

  prep_kernel<<<MROWS, 256, 0, stream>>>(input, emb, x);
  compact_kernel<<<BATCH, 256, 0, stream>>>(enc, subtok, newt);
  intent_kernel<<<BATCH, 256, 0, stream>>>(enc, sqW, sqb, iW, ib,
                                           out + (size_t)MROWS * SLOT);

  dim3 blk(128);
  dim3 gH(H / 64, MROWS / 64);     // N=768
  dim3 gF(FF / 64, MROWS / 64);    // N=512
  dim3 gS(SLOT / 64, MROWS / 64);  // N=128

  for (int l = 0; l < NL; ++l) {
    const float* Wq = saWq + (size_t)l * H * H;
    const float* Wk = saWk + (size_t)l * H * H;
    const float* Wv = saWv + (size_t)l * H * H;
    const float* Wo = saWo + (size_t)l * H * H;
    const float* sb = sab  + (size_t)l * 4 * H;
    const float* cWv = caWv + (size_t)l * H * H;
    const float* cWo = caWo + (size_t)l * H * H;
    const float* cb  = cab  + (size_t)l * 4 * H;

    // ---- self attention ----
    gemm_wmma<0><<<gH, blk, 0, stream>>>(x, Wq, sb + 0 * H, bufA, MROWS, H, H);
    gemm_wmma<0><<<gH, blk, 0, stream>>>(x, Wk, sb + 1 * H, bufB, MROWS, H, H);
    gemm_wmma<0><<<gH, blk, 0, stream>>>(x, Wv, sb + 2 * H, bufC, MROWS, H, H);
    attn_kernel<<<BATCH * NHEAD, 128, 0, stream>>>(bufA, bufB, bufC, bufD);
    gemm_wmma<0><<<gH, blk, 0, stream>>>(bufD, Wo, sb + 3 * H, bufA, MROWS, H, H);
    ln_res_kernel<<<MROWS, 256, 0, stream>>>(x, bufA,
        lng + ((size_t)l * 3 + 0) * H, lnb + ((size_t)l * 3 + 0) * H);

    // ---- cross attention: diag mask => softmax picks k==q exactly, so
    //      output = (newt @ Wv^T + bv) @ Wo^T + bo  (Q/K projections are dead) ----
    gemm_wmma<0><<<gH, blk, 0, stream>>>(newt, cWv, cb + 2 * H, bufB, MROWS, H, H);
    gemm_wmma<0><<<gH, blk, 0, stream>>>(bufB, cWo, cb + 3 * H, bufA, MROWS, H, H);
    ln_res_kernel<<<MROWS, 256, 0, stream>>>(x, bufA,
        lng + ((size_t)l * 3 + 1) * H, lnb + ((size_t)l * 3 + 1) * H);

    // ---- feed forward ----
    gemm_wmma<1><<<gF, blk, 0, stream>>>(x, fW1 + (size_t)l * FF * H,
                                         fb1 + (size_t)l * FF, bufC, MROWS, FF, H);
    gemm_wmma<0><<<gH, blk, 0, stream>>>(bufC, fW2 + (size_t)l * H * FF,
                                         fb2 + (size_t)l * H, bufA, MROWS, H, FF);
    ln_res_kernel<<<MROWS, 256, 0, stream>>>(x, bufA,
        lng + ((size_t)l * 3 + 2) * H, lnb + ((size_t)l * 3 + 2) * H);
  }

  // ---- slot head ----
  gemm_wmma<0><<<gS, blk, 0, stream>>>(x, slW, slb, bufC, MROWS, SLOT, H);
  slot_ls_kernel<<<MROWS, 128, 0, stream>>>(bufC, out);
}
